// DGCNNClassifier_66176856096775
// MI455X (gfx1250) — matrix operand order repaired and checked by reading, not compiled, not run on hardware
//
#include <hip/hip_runtime.h>
#include <hip/hip_bf16.h>

// ---------------- problem constants ----------------
#define B_    8
#define N_    2048
#define KNN_  20
#define EPS_  1e-5f

typedef __attribute__((ext_vector_type(16))) __bf16 v16bf;
typedef __attribute__((ext_vector_type(8)))  float  v8f;

// ---------------- helpers ----------------
static __device__ __forceinline__ unsigned short f32_to_bf16(float f) {
  unsigned u = __float_as_uint(f);
  u += 0x7FFFu + ((u >> 16) & 1u);           // round-to-nearest-even
  return (unsigned short)(u >> 16);
}
// Build a v16bf WMMA fragment from two contiguous 16-byte chunks.
static __device__ __forceinline__ v16bf load_frag(const unsigned short* p0,
                                                  const unsigned short* p1) {
  struct P { uint4 a, b; } p;
  p.a = *(const uint4*)p0;
  p.b = *(const uint4*)p1;
  return __builtin_bit_cast(v16bf, p);
}
static __device__ __forceinline__ void atomicMaxF(float* addr, float val) {
  int* ia = (int*)addr;
  int cur = __float_as_int(*addr);
  while (__int_as_float(cur) < val) {
    int assumed = cur;
    cur = atomicCAS(ia, assumed, __float_as_int(val));
    if (cur == assumed) break;
  }
}
static __device__ __forceinline__ float lrelu(float x) {
  return x > 0.f ? x : 0.2f * x;
}

// ---------------- per-launch prep: bf16 weights (KPAD-padded) + BN fold ----
__global__ __launch_bounds__(256) void prep_layer(
    const float* __restrict__ W, int O, int twoC, int KPAD,
    unsigned short* __restrict__ Wbf,
    const float* __restrict__ g, const float* __restrict__ b,
    const float* __restrict__ m, const float* __restrict__ v,
    float2* __restrict__ bnss) {
  int i = blockIdx.x * 256 + threadIdx.x;
  int tot = O * KPAD;
  if (i < tot) {
    int o = i / KPAD, c = i % KPAD;
    Wbf[i] = f32_to_bf16(c < twoC ? W[(size_t)o * twoC + c] : 0.f);
  }
  if (i < O) {
    float sc = g[i] * rsqrtf(v[i] + EPS_);
    bnss[i] = make_float2(sc, b[i] - m[i] * sc);
  }
}

// ---------------- kNN: per-point top-20 of (2*dot - |y|^2) ----------------
template <int C>
__global__ __launch_bounds__(256) void knn_kernel(const float* __restrict__ in,
                                                  int stride, int off,
                                                  int* __restrict__ idx) {
  const int chunks = N_ / 256;
  const int batch = blockIdx.x / chunks;
  const int n = (blockIdx.x % chunks) * 256 + threadIdx.x;

  float own[C];
  const float* base = in + ((size_t)batch * N_ + n) * stride + off;
#pragma unroll
  for (int c = 0; c < C; ++c) own[c] = base[c];

  float bestd[KNN_];
  int   besti[KNN_];
#pragma unroll
  for (int q = 0; q < KNN_; ++q) { bestd[q] = -3.0e38f; besti[q] = 0; }

  __shared__ float Feat[64][C];
  __shared__ float Sq[64];

  for (int m0 = 0; m0 < N_; m0 += 64) {
    for (int e = threadIdx.x; e < 64 * C; e += 256) {
      int r = e / C, c = e % C;
      Feat[r][c] = in[((size_t)batch * N_ + m0 + r) * stride + off + c];
    }
    __syncthreads();
    if (threadIdx.x < 64) {
      float s = 0.f;
#pragma unroll
      for (int c = 0; c < C; ++c) s += Feat[threadIdx.x][c] * Feat[threadIdx.x][c];
      Sq[threadIdx.x] = s;
    }
    __syncthreads();
#pragma unroll 1
    for (int j = 0; j < 64; ++j) {
      float dot = 0.f;
#pragma unroll
      for (int c = 0; c < C; ++c) dot += own[c] * Feat[j][c];
      float d = 2.f * dot - Sq[j];
      int m = m0 + j;
      if (d > bestd[KNN_ - 1]) {            // insert + bubble up (desc sorted)
        bestd[KNN_ - 1] = d; besti[KNN_ - 1] = m;
#pragma unroll
        for (int q = KNN_ - 1; q > 0; --q) {
          if (bestd[q] > bestd[q - 1]) {
            float td = bestd[q]; bestd[q] = bestd[q - 1]; bestd[q - 1] = td;
            int   ti = besti[q]; besti[q] = besti[q - 1]; besti[q - 1] = ti;
          }
        }
      }
    }
    __syncthreads();
  }
#pragma unroll
  for (int q = 0; q < KNN_; ++q)
    idx[((size_t)batch * N_ + n) * KNN_ + q] = besti[q];
}

// ---------------- fused EdgeConv: gather + bf16 WMMA + BN + LReLU + max_k ----
// 4 points per workgroup -> 80 edge rows = 5 row-tiles of 16.
template <int CIN, int O>
__global__ __launch_bounds__(256) void edgeconv_kernel(
    const float* __restrict__ in, int inStride, int inOff,
    const int* __restrict__ idx,
    const unsigned short* __restrict__ Wbf,   // (O, KPAD) bf16, zero-padded
    const float2* __restrict__ bnss,          // (O) folded BN scale/shift
    float* __restrict__ out, int outStride, int outOff) {
  constexpr int TWO_C = 2 * CIN;
  constexpr int KPAD = ((TWO_C + 31) / 32) * 32;
  constexpr int R = 4 * KNN_;          // 80 edge rows
  constexpr int RT = R / 16;           // 5 row tiles
  constexpr int CT = O / 16;           // col tiles
  constexpr int KS = KPAD / 32;        // bf16 K-steps

  alignas(16) __shared__ unsigned short Afeat[R][KPAD];
  __shared__ float OutLds[4][O];

  const int pt0 = blockIdx.x * 4;
  const int batch = pt0 / N_;
  const int n0 = pt0 % N_;

  for (int e = threadIdx.x; e < 4 * O; e += 256) ((float*)OutLds)[e] = -3.0e38f;

  // stage edge features [nbr-ctr | ctr] as bf16, zero-padded to KPAD
  for (int e = threadIdx.x; e < R * KPAD; e += 256) {
    int row = e / KPAD, c = e % KPAD;
    int p = row / KNN_, kk = row % KNN_;
    int n = n0 + p;
    float val = 0.f;
    if (c < TWO_C) {
      if (c < CIN) {
        int nb = idx[((size_t)batch * N_ + n) * KNN_ + kk];
        float nv = in[((size_t)batch * N_ + nb) * inStride + inOff + c];
        float cv = in[((size_t)batch * N_ + n) * inStride + inOff + c];
        val = nv - cv;
      } else {
        val = in[((size_t)batch * N_ + n) * inStride + inOff + (c - CIN)];
      }
    }
    Afeat[row][c] = f32_to_bf16(val);
  }
  __syncthreads();

  const int lane = threadIdx.x & 31;
  const int wid = threadIdx.x >> 5;
  const int mrow = lane & 15;
  const int hi = lane >> 4;

  for (int job = wid; job < RT * CT; job += 8) {
    const int rt = job % RT, ct = job / RT;
    const int ocol = ct * 16 + mrow;
    const unsigned short* wr = Wbf + (size_t)ocol * KPAD;
    const unsigned short* arow = &Afeat[rt * 16 + mrow][hi * 8];
    v8f acc = {};
#pragma unroll 1
    for (int ks = 0; ks < KS; ++ks) {
      const int k0 = ks * 32;
      const int c0 = k0 + hi * 8;
      // A fragment: 16-bit 16x32 layout -> two contiguous 16B LDS chunks
      v16bf a = load_frag(arow + k0, arow + k0 + 16);
      // B fragment: 32x16 -> two 16B chunks per lane (same k order as A) from bf16 weights
      v16bf bf = load_frag(wr + c0, wr + c0 + 16);
      __builtin_prefetch(wr + c0 + 32, 0, 1);   // next K-step (global_prefetch)
      acc = __builtin_amdgcn_wmma_f32_16x16x32_bf16(false, a, false, bf,
                                                    (short)0, acc, false, false);
    }
    const float2 ss = bnss[ocol];
#pragma unroll
    for (int i = 0; i < 8; ++i) {
      int row = rt * 16 + i + 8 * hi;
      int p = row / KNN_;
      atomicMaxF(&OutLds[p][ocol], lrelu(acc[i] * ss.x + ss.y));
    }
  }
  __syncthreads();

  for (int e = threadIdx.x; e < 4 * O; e += 256) {
    int p = e / O, o = e % O;
    out[((size_t)batch * N_ + n0 + p) * outStride + outOff + o] = OutLds[p][o];
  }
}

// ---------------- embedding GEMM (512->1024) + BN/LReLU + pooled max/sum ----
__global__ __launch_bounds__(256) void embed_kernel(
    const float* __restrict__ xc,
    const unsigned short* __restrict__ Wbf,   // (1024, 512) bf16
    const float2* __restrict__ bnss,          // (1024)
    float* __restrict__ pmax, float* __restrict__ psum) {
  const int tiles = N_ / 16;
  const int batch = blockIdx.x / tiles;
  const int n0 = (blockIdx.x % tiles) * 16;

  alignas(16) __shared__ unsigned short Ain[16][512];
  for (int e = threadIdx.x; e < 16 * 512; e += 256) {
    int r = e / 512, c = e % 512;
    Ain[r][c] = f32_to_bf16(xc[((size_t)batch * N_ + n0 + r) * 512 + c]);
  }
  __syncthreads();

  const int lane = threadIdx.x & 31;
  const int wid = threadIdx.x >> 5;
  const int mrow = lane & 15;
  const int hi = lane >> 4;

  for (int ct = wid; ct < 64; ct += 8) {
    const int ocol = ct * 16 + mrow;
    const unsigned short* wr = Wbf + (size_t)ocol * 512;
    const unsigned short* arow = &Ain[mrow][hi * 8];
    v8f acc = {};
#pragma unroll 1
    for (int ks = 0; ks < 16; ++ks) {
      const int k0 = ks * 32;
      const int c0 = k0 + hi * 8;
      v16bf a = load_frag(arow + k0, arow + k0 + 16);
      v16bf bf = load_frag(wr + c0, wr + c0 + 16);
      __builtin_prefetch(wr + c0 + 32, 0, 1);
      acc = __builtin_amdgcn_wmma_f32_16x16x32_bf16(false, a, false, bf,
                                                    (short)0, acc, false, false);
    }
    const float2 ss = bnss[ocol];
    float pm = -3.0e38f, ps = 0.f;
#pragma unroll
    for (int i = 0; i < 8; ++i) {
      float y = lrelu(acc[i] * ss.x + ss.y);
      pm = fmaxf(pm, y);
      ps += y;
    }
    // fold the two half-tiles (rows 0-7 vs 8-15 live in lane pairs l, l^16)
    pm = fmaxf(pm, __shfl_xor(pm, 16, 32));
    ps += __shfl_xor(ps, 16, 32);
    if (hi == 0) {
      atomicMaxF(&pmax[batch * 1024 + ocol], pm);
      atomicAdd(&psum[batch * 1024 + ocol], ps);
    }
  }
}

// ---------------- pooled-buffer init ----------------
__global__ void init_pool(float* pmax, float* psum, int n) {
  int i = blockIdx.x * 256 + threadIdx.x;
  if (i < n) { pmax[i] = -3.0e38f; psum[i] = 0.f; }
}

// ---------------- classifier head (tiny) ----------------
__global__ __launch_bounds__(256) void head_kernel(
    const float* __restrict__ pmax, const float* __restrict__ psum,
    const float* __restrict__ fc1_w,
    const float* __restrict__ g6, const float* __restrict__ b6,
    const float* __restrict__ m6, const float* __restrict__ v6,
    const float* __restrict__ fc2_w, const float* __restrict__ fc2_b,
    const float* __restrict__ g7, const float* __restrict__ b7,
    const float* __restrict__ m7, const float* __restrict__ v7,
    const float* __restrict__ fc3_w, const float* __restrict__ fc3_b,
    float* __restrict__ out) {
  const int b = blockIdx.x;
  const int tid = threadIdx.x;
  __shared__ float hin[2048];
  __shared__ float h1[512];
  __shared__ float h2[256];

  for (int c = tid; c < 1024; c += 256) {
    hin[c] = pmax[b * 1024 + c];
    hin[1024 + c] = psum[b * 1024 + c] * (1.0f / (float)N_);
  }
  __syncthreads();

  for (int o = tid; o < 512; o += 256) {
    const float* wr = fc1_w + (size_t)o * 2048;
    float s = 0.f;
    for (int c = 0; c < 2048; ++c) s += hin[c] * wr[c];
    float scale = g6[o] * rsqrtf(v6[o] + EPS_);
    h1[o] = lrelu(s * scale + (b6[o] - m6[o] * scale));
  }
  __syncthreads();

  for (int o = tid; o < 256; o += 256) {
    const float* wr = fc2_w + (size_t)o * 512;
    float s = fc2_b[o];
    for (int c = 0; c < 512; ++c) s += h1[c] * wr[c];
    float scale = g7[o] * rsqrtf(v7[o] + EPS_);
    h2[o] = lrelu(s * scale + (b7[o] - m7[o] * scale));
  }
  __syncthreads();

  if (tid < 10) {
    const float* wr = fc3_w + (size_t)tid * 256;
    float s = fc3_b[tid];
    for (int c = 0; c < 256; ++c) s += h2[c] * wr[c];
    out[b * 10 + tid] = s;
  }
}

// ---------------- launcher ----------------
extern "C" void kernel_launch(void* const* d_in, const int* in_sizes, int n_in,
                              void* d_out, int out_size, void* d_ws, size_t ws_size,
                              hipStream_t stream) {
  (void)in_sizes; (void)n_in; (void)out_size; (void)ws_size;
  const float* x = (const float*)d_in[0];
  const float *w1 = (const float*)d_in[1],  *g1 = (const float*)d_in[2],
              *b1 = (const float*)d_in[3],  *m1 = (const float*)d_in[4],  *v1 = (const float*)d_in[5];
  const float *w2 = (const float*)d_in[6],  *g2 = (const float*)d_in[7],
              *b2 = (const float*)d_in[8],  *m2 = (const float*)d_in[9],  *v2 = (const float*)d_in[10];
  const float *w3 = (const float*)d_in[11], *g3 = (const float*)d_in[12],
              *b3 = (const float*)d_in[13], *m3 = (const float*)d_in[14], *v3 = (const float*)d_in[15];
  const float *w4 = (const float*)d_in[16], *g4 = (const float*)d_in[17],
              *b4 = (const float*)d_in[18], *m4 = (const float*)d_in[19], *v4 = (const float*)d_in[20];
  const float *w5 = (const float*)d_in[21], *g5 = (const float*)d_in[22],
              *b5 = (const float*)d_in[23], *m5 = (const float*)d_in[24], *v5 = (const float*)d_in[25];
  const float *fc1_w = (const float*)d_in[26];
  const float *g6 = (const float*)d_in[27], *b6 = (const float*)d_in[28],
              *m6 = (const float*)d_in[29], *v6 = (const float*)d_in[30];
  const float *fc2_w = (const float*)d_in[31], *fc2_b = (const float*)d_in[32];
  const float *g7 = (const float*)d_in[33], *b7 = (const float*)d_in[34],
              *m7 = (const float*)d_in[35], *v7 = (const float*)d_in[36];
  const float *fc3_w = (const float*)d_in[37], *fc3_b = (const float*)d_in[38];
  float* out = (float*)d_out;

  char* wsb = (char*)d_ws;
  int*   idxbuf = (int*)wsb;                                  // B*N*K ints (1.31 MB)
  float* concat = (float*)(wsb + ((size_t)2 << 20));          // B*N*512 f32 (33.6 MB)
  float* pmax   = (float*)(wsb + ((size_t)36 << 20));         // B*1024
  float* psum   = pmax + B_ * 1024;                           // B*1024
  // bf16 weight buffers, rows padded to KPAD
  unsigned short* w1bf = (unsigned short*)(wsb + ((size_t)37 << 20)); // 64*32
  unsigned short* w2bf = w1bf + 64 * 32;                              // 64*128
  unsigned short* w3bf = w2bf + 64 * 128;                             // 128*128
  unsigned short* w4bf = w3bf + 128 * 128;                            // 256*256
  unsigned short* w5bf = w4bf + 256 * 256;                            // 1024*512
  float2* s1 = (float2*)(wsb + ((size_t)39 << 20));                   // 64
  float2* s2 = s1 + 64;                                               // 64
  float2* s3 = s2 + 64;                                               // 128
  float2* s4 = s3 + 128;                                              // 256
  float2* s5 = s4 + 256;                                              // 1024

  const dim3 blk(256);
  const dim3 gKnn(B_ * (N_ / 256));
  const dim3 gEc(B_ * N_ / 4);
  const dim3 gEmb(B_ * (N_ / 16));

  // ---- per-launch prep: bf16 weights + folded BN ----
  prep_layer<<<dim3((64 * 32 + 255) / 256), blk, 0, stream>>>(w1, 64, 6, 32, w1bf, g1, b1, m1, v1, s1);
  prep_layer<<<dim3((64 * 128 + 255) / 256), blk, 0, stream>>>(w2, 64, 128, 128, w2bf, g2, b2, m2, v2, s2);
  prep_layer<<<dim3((128 * 128 + 255) / 256), blk, 0, stream>>>(w3, 128, 128, 128, w3bf, g3, b3, m3, v3, s3);
  prep_layer<<<dim3((256 * 256 + 255) / 256), blk, 0, stream>>>(w4, 256, 256, 256, w4bf, g4, b4, m4, v4, s4);
  prep_layer<<<dim3((1024 * 512 + 255) / 256), blk, 0, stream>>>(w5, 1024, 512, 512, w5bf, g5, b5, m5, v5, s5);

  // EdgeConv 1: x(B,N,3) -> concat[:, 0:64]
  knn_kernel<3><<<gKnn, blk, 0, stream>>>(x, 3, 0, idxbuf);
  edgeconv_kernel<3, 64><<<gEc, blk, 0, stream>>>(x, 3, 0, idxbuf, w1bf, s1, concat, 512, 0);
  // EdgeConv 2: concat[:,0:64] -> concat[:, 64:128]
  knn_kernel<64><<<gKnn, blk, 0, stream>>>(concat, 512, 0, idxbuf);
  edgeconv_kernel<64, 64><<<gEc, blk, 0, stream>>>(concat, 512, 0, idxbuf, w2bf, s2, concat, 512, 64);
  // EdgeConv 3: concat[:,64:128] -> concat[:, 128:256]
  knn_kernel<64><<<gKnn, blk, 0, stream>>>(concat, 512, 64, idxbuf);
  edgeconv_kernel<64, 128><<<gEc, blk, 0, stream>>>(concat, 512, 64, idxbuf, w3bf, s3, concat, 512, 128);
  // EdgeConv 4: concat[:,128:256] -> concat[:, 256:512]
  knn_kernel<128><<<gKnn, blk, 0, stream>>>(concat, 512, 128, idxbuf);
  edgeconv_kernel<128, 256><<<gEc, blk, 0, stream>>>(concat, 512, 128, idxbuf, w4bf, s4, concat, 512, 256);
  // Embedding + global pooling
  init_pool<<<dim3((B_ * 1024 + 255) / 256), blk, 0, stream>>>(pmax, psum, B_ * 1024);
  embed_kernel<<<gEmb, blk, 0, stream>>>(concat, w5bf, s5, pmax, psum);
  // Head
  head_kernel<<<dim3(B_), blk, 0, stream>>>(pmax, psum, fc1_w, g6, b6, m6, v6,
      fc2_w, fc2_b, g7, b7, m7, v7, fc3_w, fc3_b, out);
}
